// MDTA_3384434229420
// MI455X (gfx1250) — matrix operand order repaired and checked by reading, not compiled, not run on hardware
//
#include <hip/hip_runtime.h>
#include <stdint.h>

#define DIMC 192
#define HEADS 4
#define HD 48
#define HDP 64          // head_dim padded to 64 for two K=32 WMMA steps
#define HGT 48
#define WID 48
#define NTOK 2304       // 48*48
#define NB 2

typedef __attribute__((ext_vector_type(16))) __bf16 v16bf;
typedef __attribute__((ext_vector_type(8)))  float  v8f;

union FragAB { v16bf v; unsigned int u[8]; };
union FragC  { v8f   v; float        f[8]; };

__device__ __forceinline__ unsigned short f2bf(float f) {
  unsigned int u = __builtin_bit_cast(unsigned int, f);
  u += 0x7FFFu + ((u >> 16) & 1u);          // round-to-nearest-even
  return (unsigned short)(u >> 16);
}

// A-fragment (16x32 bf16) K-start for VGPR r, lane-half h (contiguous: k = 16h + 2r)
__device__ __forceinline__ int a_k0(int r, int h) {
  return 2 * r + 16 * h;
}

__device__ __forceinline__ v8f wmma_bf16(v16bf a, v16bf b, v8f c) {
  return __builtin_amdgcn_wmma_f32_16x16x32_bf16(false, a, false, b, (short)0, c,
                                                 false, false);
}

__device__ __forceinline__ unsigned int ld32(const unsigned short* p) {
  return *(const unsigned int*)p;
}

// ---------------- prep: x (B,C,H,W) f32 -> xT (B,N,C) bf16 -------------------
__global__ void k_prep_x(const float* __restrict__ x, unsigned short* __restrict__ xT) {
  int idx = blockIdx.x * blockDim.x + threadIdx.x;
  if (idx >= NB * NTOK * DIMC) return;
  int c = idx % DIMC;
  int n = (idx / DIMC) % NTOK;
  int b = idx / (DIMC * NTOK);
  xT[idx] = f2bf(x[((size_t)b * DIMC + c) * NTOK + n]);
}

// ---------------- prep: weight (192,192) f32 -> bf16 -------------------------
__global__ void k_w2bf(const float* __restrict__ w, unsigned short* __restrict__ wb) {
  int idx = blockIdx.x * blockDim.x + threadIdx.x;
  if (idx >= DIMC * DIMC) return;
  wb[idx] = f2bf(w[idx]);
}

// ---------------- fused QKV pointwise GEMM (WMMA bf16) -----------------------
// Y[oc,n] = sum_c W[oc,c] * X[c,n], X given token-major (n,c).
// one wave = one 16x16 output tile, 3 accumulators (q,k,v).
__global__ void __launch_bounds__(256)
k_pw_qkv(const unsigned short* __restrict__ xT,
         const unsigned short* __restrict__ wq,
         const unsigned short* __restrict__ wk,
         const unsigned short* __restrict__ wv,
         float* __restrict__ Qp, float* __restrict__ Kp, float* __restrict__ Vp) {
  int wave = (blockIdx.x * blockDim.x + threadIdx.x) >> 5;  // 0 .. 3455
  int lane = threadIdx.x & 31;
  int half = lane >> 4, l16 = lane & 15;
  int nt = wave % (NTOK / 16);
  int mt = (wave / (NTOK / 16)) % (DIMC / 16);
  int b  = wave / ((NTOK / 16) * (DIMC / 16));

  FragC q{}, k{}, v{};
  const unsigned short* xrow = xT + ((size_t)b * NTOK + nt * 16 + l16) * DIMC;
  int arow = mt * 16 + l16;

  for (int kc = 0; kc < DIMC / 32; ++kc) {
    FragAB bx, aq, ak, av;
    int kb = kc * 32 + half * 16;
#pragma unroll
    for (int r = 0; r < 8; ++r) bx.u[r] = ld32(xrow + kb + 2 * r);
#pragma unroll
    for (int r = 0; r < 8; ++r) {
      int k0 = kc * 32 + a_k0(r, half);
      aq.u[r] = ld32(wq + (size_t)arow * DIMC + k0);
      ak.u[r] = ld32(wk + (size_t)arow * DIMC + k0);
      av.u[r] = ld32(wv + (size_t)arow * DIMC + k0);
    }
    q.v = wmma_bf16(aq.v, bx.v, q.v);
    k.v = wmma_bf16(ak.v, bx.v, k.v);
    v.v = wmma_bf16(av.v, bx.v, v.v);
  }
#pragma unroll
  for (int r = 0; r < 8; ++r) {
    int oc = mt * 16 + r + 8 * half;
    int n  = nt * 16 + l16;
    size_t o = ((size_t)b * DIMC + oc) * NTOK + n;
    Qp[o] = q.f[r]; Kp[o] = k.f[r]; Vp[o] = v.f[r];
  }
}

// ------------- depthwise 3x3 -> token-major (B,H,N,64) bf16, zero-padded -----
__global__ void k_dw_qk(const float* __restrict__ in, const float* __restrict__ w,
                        unsigned short* __restrict__ out) {
  int idx = blockIdx.x * blockDim.x + threadIdx.x;
  if (idx >= NB * HEADS * NTOK * HDP) return;
  int d = idx % HDP;
  int n = (idx / HDP) % NTOK;
  int h = (idx / (HDP * NTOK)) % HEADS;
  int b = idx / (HDP * NTOK * HEADS);
  if (d >= HD) { out[idx] = 0; return; }
  int c = h * HD + d;
  int y = n / WID, xx = n % WID;
  float s = 0.f;
#pragma unroll
  for (int ky = 0; ky < 3; ++ky) {
#pragma unroll
    for (int kx = 0; kx < 3; ++kx) {
      int yy = y + ky - 1, xc = xx + kx - 1;
      if (yy >= 0 && yy < HGT && xc >= 0 && xc < WID)
        s += in[((size_t)b * DIMC + c) * NTOK + yy * WID + xc] * w[c * 9 + ky * 3 + kx];
    }
  }
  out[idx] = f2bf(s);
}

// ------------- depthwise 3x3 -> channel-major (B,C,N) bf16 (for V) -----------
__global__ void k_dw_v(const float* __restrict__ in, const float* __restrict__ w,
                       unsigned short* __restrict__ out) {
  int idx = blockIdx.x * blockDim.x + threadIdx.x;
  if (idx >= NB * DIMC * NTOK) return;
  int n = idx % NTOK;
  int c = (idx / NTOK) % DIMC;
  int b = idx / (NTOK * DIMC);
  int y = n / WID, xx = n % WID;
  float s = 0.f;
#pragma unroll
  for (int ky = 0; ky < 3; ++ky) {
#pragma unroll
    for (int kx = 0; kx < 3; ++kx) {
      int yy = y + ky - 1, xc = xx + kx - 1;
      if (yy >= 0 && yy < HGT && xc >= 0 && xc < WID)
        s += in[((size_t)b * DIMC + c) * NTOK + yy * WID + xc] * w[c * 9 + ky * 3 + kx];
    }
  }
  out[idx] = f2bf(s);
}

// ---------------- flash attention (WMMA bf16, online softmax) ----------------
// One wave: 16 query rows, stream keys in 32-col chunks. 8 waves / block.
__global__ void __launch_bounds__(256)
k_attn(const unsigned short* __restrict__ Qt,   // (B,H,N,64) bf16
       const unsigned short* __restrict__ Kt,   // (B,H,N,64) bf16
       const unsigned short* __restrict__ Vt,   // (B,C,N)    bf16  (= B,H,48,N)
       const float* __restrict__ scale,
       unsigned short* __restrict__ Ot) {       // (B,N,C)    bf16
  __shared__ unsigned short Pld[8 * 16 * 32];
  int wv = threadIdx.x >> 5;
  int lane = threadIdx.x & 31;
  int half = lane >> 4, l16 = lane & 15;
  int gw = blockIdx.x * 8 + wv;                 // 0 .. 1151
  int nt = gw % (NTOK / 16);
  int h  = (gw / (NTOK / 16)) % HEADS;
  int b  = gw / ((NTOK / 16) * HEADS);
  int bh = b * HEADS + h;
  float inv = 1.0f / scale[0];

  // Q A-fragments, d = 0..63 in two K=32 chunks; resident all iterations
  FragAB qa[2];
  const unsigned short* qp = Qt + ((size_t)bh * NTOK + nt * 16 + l16) * HDP;
#pragma unroll
  for (int kc = 0; kc < 2; ++kc)
#pragma unroll
    for (int r = 0; r < 8; ++r)
      qa[kc].u[r] = ld32(qp + kc * 32 + a_k0(r, half));

  float mrow[8], lrow[8];
  FragC o0{}, o1{}, o2{};
#pragma unroll
  for (int r = 0; r < 8; ++r) { mrow[r] = -1e30f; lrow[r] = 0.f; }

  unsigned short* P = Pld + wv * (16 * 32);

  for (int m0 = 0; m0 < NTOK; m0 += 32) {
    // ---- S = Q K^T for 32 key columns (two 16-col halves) ----
    FragC sL{}, sR{};
#pragma unroll
    for (int kc = 0; kc < 2; ++kc) {
      FragAB kbL, kbR;
      int kb = kc * 32 + half * 16;
      const unsigned short* kpL = Kt + ((size_t)bh * NTOK + m0 + l16) * HDP + kb;
      const unsigned short* kpR = kpL + 16 * HDP;
#pragma unroll
      for (int r = 0; r < 8; ++r) { kbL.u[r] = ld32(kpL + 2 * r); kbR.u[r] = ld32(kpR + 2 * r); }
      sL.v = wmma_bf16(qa[kc].v, kbL.v, sL.v);
      sR.v = wmma_bf16(qa[kc].v, kbR.v, sR.v);
    }
    // ---- online softmax on D-layout (row = r + 8*half, cols across 16 lanes)
    float alpha[8];
#pragma unroll
    for (int r = 0; r < 8; ++r) {
      float a = sL.f[r] * inv, bb = sR.f[r] * inv;
      float mx = fmaxf(a, bb);
#pragma unroll
      for (int off = 8; off > 0; off >>= 1) mx = fmaxf(mx, __shfl_xor(mx, off, 16));
      float mn = fmaxf(mrow[r], mx);
      alpha[r] = __expf(mrow[r] - mn);
      mrow[r] = mn;
      float pa = __expf(a - mn), pb = __expf(bb - mn);
      sL.f[r] = pa; sR.f[r] = pb;
      float rs = pa + pb;
#pragma unroll
      for (int off = 8; off > 0; off >>= 1) rs += __shfl_xor(rs, off, 16);
      lrow[r] = lrow[r] * alpha[r] + rs;
    }
#pragma unroll
    for (int r = 0; r < 8; ++r) { o0.f[r] *= alpha[r]; o1.f[r] *= alpha[r]; o2.f[r] *= alpha[r]; }

    // ---- P: D-layout -> LDS (row-major 16x32) -> A-fragment ----
#pragma unroll
    for (int r = 0; r < 8; ++r) {
      int row = r + 8 * half;
      P[row * 32 + l16]      = f2bf(sL.f[r]);
      P[row * 32 + 16 + l16] = f2bf(sR.f[r]);
    }
    FragAB pf;
#pragma unroll
    for (int r = 0; r < 8; ++r)
      pf.u[r] = ld32(P + l16 * 32 + a_k0(r, half));

    // ---- O += P @ V for three 16-wide d-tiles ----
    {
      const unsigned short* vbase =
          Vt + ((size_t)b * DIMC + h * HD + l16) * NTOK + m0 + half * 16;
      FragAB vb;
#pragma unroll
      for (int r = 0; r < 8; ++r) vb.u[r] = ld32(vbase + 2 * r);
      o0.v = wmma_bf16(pf.v, vb.v, o0.v);
#pragma unroll
      for (int r = 0; r < 8; ++r) vb.u[r] = ld32(vbase + 16 * NTOK + 2 * r);
      o1.v = wmma_bf16(pf.v, vb.v, o1.v);
#pragma unroll
      for (int r = 0; r < 8; ++r) vb.u[r] = ld32(vbase + 32 * NTOK + 2 * r);
      o2.v = wmma_bf16(pf.v, vb.v, o2.v);
    }
  }

  // ---- epilogue: normalize, store token-major (B,N,C) bf16 ----
#pragma unroll
  for (int r = 0; r < 8; ++r) {
    int n = nt * 16 + r + 8 * half;
    float rl = 1.0f / lrow[r];
    size_t base = ((size_t)b * NTOK + n) * DIMC + h * HD;
    Ot[base + l16]      = f2bf(o0.f[r] * rl);
    Ot[base + 16 + l16] = f2bf(o1.f[r] * rl);
    Ot[base + 32 + l16] = f2bf(o2.f[r] * rl);
  }
}

// ---------------- output pointwise GEMM + residual ---------------------------
__global__ void __launch_bounds__(256)
k_out(const unsigned short* __restrict__ Ot,    // (B,N,C) bf16
      const unsigned short* __restrict__ wo,    // (192,192) bf16
      const float* __restrict__ xres,           // (B,C,N) f32
      float* __restrict__ out) {                // (B,C,N) f32
  int wave = (blockIdx.x * blockDim.x + threadIdx.x) >> 5;
  int lane = threadIdx.x & 31;
  int half = lane >> 4, l16 = lane & 15;
  int nt = wave % (NTOK / 16);
  int mt = (wave / (NTOK / 16)) % (DIMC / 16);
  int b  = wave / ((NTOK / 16) * (DIMC / 16));

  FragC acc{};
  const unsigned short* orow = Ot + ((size_t)b * NTOK + nt * 16 + l16) * DIMC;
  int arow = mt * 16 + l16;
  for (int kc = 0; kc < DIMC / 32; ++kc) {
    FragAB bx, aw;
    int kb = kc * 32 + half * 16;
#pragma unroll
    for (int r = 0; r < 8; ++r) bx.u[r] = ld32(orow + kb + 2 * r);
#pragma unroll
    for (int r = 0; r < 8; ++r)
      aw.u[r] = ld32(wo + (size_t)arow * DIMC + kc * 32 + a_k0(r, half));
    acc.v = wmma_bf16(aw.v, bx.v, acc.v);
  }
#pragma unroll
  for (int r = 0; r < 8; ++r) {
    int oc = mt * 16 + r + 8 * half;
    int n  = nt * 16 + l16;
    size_t o = ((size_t)b * DIMC + oc) * NTOK + n;
    out[o] = acc.f[r] + xres[o];
  }
}

// -----------------------------------------------------------------------------
extern "C" void kernel_launch(void* const* d_in, const int* in_sizes, int n_in,
                              void* d_out, int out_size, void* d_ws, size_t ws_size,
                              hipStream_t stream) {
  (void)in_sizes; (void)n_in; (void)out_size; (void)ws_size;
  const float* x     = (const float*)d_in[0];
  const float* q_pw  = (const float*)d_in[1];
  const float* q_dw  = (const float*)d_in[2];
  const float* k_pw  = (const float*)d_in[3];
  const float* k_dw  = (const float*)d_in[4];
  const float* v_pw  = (const float*)d_in[5];
  const float* v_dw  = (const float*)d_in[6];
  const float* scale = (const float*)d_in[7];
  const float* o_pw  = (const float*)d_in[8];
  float* out = (float*)d_out;

  char* base = (char*)d_ws;
  size_t off = 0;
  auto alloc = [&](size_t bytes) -> char* {
    char* r = base + off;
    off = (off + bytes + 255) & ~(size_t)255;
    return r;
  };

  unsigned short* xT = (unsigned short*)alloc((size_t)NB * NTOK * DIMC * 2);
  unsigned short* wq = (unsigned short*)alloc((size_t)DIMC * DIMC * 2);
  unsigned short* wk = (unsigned short*)alloc((size_t)DIMC * DIMC * 2);
  unsigned short* wv = (unsigned short*)alloc((size_t)DIMC * DIMC * 2);
  unsigned short* wo = (unsigned short*)alloc((size_t)DIMC * DIMC * 2);
  float* Qp = (float*)alloc((size_t)NB * DIMC * NTOK * 4);
  float* Kp = (float*)alloc((size_t)NB * DIMC * NTOK * 4);
  float* Vp = (float*)alloc((size_t)NB * DIMC * NTOK * 4);
  unsigned short* Qt = (unsigned short*)alloc((size_t)NB * HEADS * NTOK * HDP * 2);
  unsigned short* Kt = (unsigned short*)alloc((size_t)NB * HEADS * NTOK * HDP * 2);
  unsigned short* Vt = (unsigned short*)alloc((size_t)NB * DIMC * NTOK * 2);
  unsigned short* Ot = (unsigned short*)alloc((size_t)NB * NTOK * DIMC * 2);

  k_prep_x<<<(NB * NTOK * DIMC + 255) / 256, 256, 0, stream>>>(x, xT);
  k_w2bf<<<(DIMC * DIMC + 255) / 256, 256, 0, stream>>>(q_pw, wq);
  k_w2bf<<<(DIMC * DIMC + 255) / 256, 256, 0, stream>>>(k_pw, wk);
  k_w2bf<<<(DIMC * DIMC + 255) / 256, 256, 0, stream>>>(v_pw, wv);
  k_w2bf<<<(DIMC * DIMC + 255) / 256, 256, 0, stream>>>(o_pw, wo);

  // 2 * 12 * 144 = 3456 wave-tiles, 8 waves / block
  k_pw_qkv<<<3456 / 8, 256, 0, stream>>>(xT, wq, wk, wv, Qp, Kp, Vp);

  k_dw_qk<<<(NB * HEADS * NTOK * HDP + 255) / 256, 256, 0, stream>>>(Qp, q_dw, Qt);
  k_dw_qk<<<(NB * HEADS * NTOK * HDP + 255) / 256, 256, 0, stream>>>(Kp, k_dw, Kt);
  k_dw_v<<<(NB * DIMC * NTOK + 255) / 256, 256, 0, stream>>>(Vp, v_dw, Vt);

  // 2 * 4 * 144 = 1152 wave-tiles, 8 waves / block
  k_attn<<<1152 / 8, 256, 0, stream>>>(Qt, Kt, Vt, scale, Ot);

  k_out<<<3456 / 8, 256, 0, stream>>>(Ot, wo, x, out);
}
